// Block_16192026705931
// MI455X (gfx1250) — matrix operand order repaired and checked by reading, not compiled, not run on hardware
//
#include <hip/hip_runtime.h>
#include <hip/hip_bf16.h>

// ---------------- problem constants ----------------
#define BATCH   2
#define SEQ     2048
#define NTOK    (BATCH * SEQ)      // 4096
#define DMODEL  1024
#define NHEAD   16
#define HEADD   64
#define NEXP    8
#define FDIM    4096
#define EPS     1e-5f
#define TWO_PI  6.28318530717958647692f

typedef __attribute__((ext_vector_type(16))) __bf16 v16bf;
typedef __attribute__((ext_vector_type(8)))  __bf16 v8bf;
typedef __attribute__((ext_vector_type(4)))  __bf16 v4bf;
typedef __attribute__((ext_vector_type(2)))  __bf16 v2bf;
typedef __attribute__((ext_vector_type(8)))  float  v8f;
typedef __attribute__((ext_vector_type(4)))  float  v4f;

// ---------------- async global->LDS (ASYNCcnt) with graceful fallback ------
#if defined(__gfx1250__) &&                                        \
    __has_builtin(__builtin_amdgcn_global_load_async_to_lds_b128) && \
    __has_builtin(__builtin_amdgcn_s_wait_asynccnt)
#define HAVE_ASYNC 1
#else
#define HAVE_ASYNC 0
#endif

#if HAVE_ASYNC
// Builtin parameter type per hipcc diagnostic: pointer to 16B int vector.
typedef int v4i_t __attribute__((vector_size(16)));
typedef __attribute__((address_space(1))) v4i_t g_v4i;
typedef __attribute__((address_space(3))) v4i_t l_v4i;
#endif

// Copy 16B global -> LDS. Async DMA when available (no VGPR round-trip),
// else plain vector copy.
__device__ __forceinline__ void cp16_g2l(const __bf16* g, __bf16* l) {
#if HAVE_ASYNC
  __builtin_amdgcn_global_load_async_to_lds_b128((g_v4i*)g, (l_v4i*)l, 0, 0);
#else
  *(v4f*)l = *(const v4f*)g;
#endif
}

// Wait for all outstanding async LDS copies of this wave.
__device__ __forceinline__ void async_fence() {
#if HAVE_ASYNC
  __builtin_amdgcn_s_wait_asynccnt(0);
#endif
}

// ---------------- WMMA wrapper ----------------
__device__ __forceinline__ v8f wmma_bf16(v16bf a, v16bf b, v8f c) {
  // D = A(16x32 bf16) * B(32x16 bf16) + C(16x16 f32)
  return __builtin_amdgcn_wmma_f32_16x16x32_bf16(
      /*neg_a=*/false, a, /*neg_b=*/false, b,
      /*c_mod=*/(short)0, c, /*reuse_a=*/false, /*reuse_b=*/false);
}

// Build a 16-element bf16 fragment from two contiguous 8-element LDS chunks.
__device__ __forceinline__ v16bf ld_frag(const __bf16* p0, const __bf16* p1) {
  v8bf lo = *(const v8bf*)p0;
  v8bf hi = *(const v8bf*)p1;
  v16bf r;
#pragma unroll
  for (int i = 0; i < 8; ++i) { r[i] = lo[i]; r[i + 8] = hi[i]; }
  return r;
}

// ---------------- GEMM tile machinery ----------------
// Block tile: 128(M) x 64(N), K-step 32, double-buffered LDS.
// 8 waves, each wave: 16 rows x 64 cols = 4 C-fragments (v8f).
// As[buf][128][32] row-major; Bs[buf][64][32] stored [n][k] so each lane's
// B fragment (16 contiguous K for one column) is two 16B ds_load_b128.

// A tile (bf16, layout-preserving): async DMA, 2x b128 per thread.
__device__ __forceinline__ void load_a_tile(const __bf16* __restrict__ A, int lda,
                                            int m0, int k0, __bf16 (*As)[32]) {
  int t = threadIdx.x;          // 256 threads x 16 elems = 128*32
  int r = t >> 1;
  int c = (t & 1) * 16;
  const __bf16* src = A + (size_t)(m0 + r) * lda + k0 + c;
  cp16_g2l(src,     &As[r][c]);
  cp16_g2l(src + 8, &As[r][c + 8]);
}

// B row-major [K][N] (f32) -> bf16, transposed into Bs[n][k].
// Each thread: 2 consecutive k rows x 4 n -> pack bf16 pairs along k,
// 4x ds_store_b32.
__device__ __forceinline__ void load_b_tile_n(const float* __restrict__ B, int ldb,
                                              int k0, int n0, __bf16 (*Bs)[32]) {
  int t = threadIdx.x;
  int k = (t >> 4) * 2;         // 0,2,...,30
  int nc = (t & 15) * 4;        // 0,4,...,60
  const float* s0 = B + (size_t)(k0 + k) * ldb + n0 + nc;
  const float* s1 = s0 + ldb;
  v4f a = *(const v4f*)s0;
  v4f b = *(const v4f*)s1;
#pragma unroll
  for (int i = 0; i < 4; ++i) {
    v2bf p;
    p[0] = (__bf16)a[i];
    p[1] = (__bf16)b[i];
    *(v2bf*)&Bs[nc + i][k] = p;
  }
}

// B accessed transposed: B[k][n] = src[n*ldb + k] (e.g. w_proj.T).
// 8 converted values are contiguous in k -> one ds_store_b128.
__device__ __forceinline__ void load_b_tile_t(const float* __restrict__ B, int ldb,
                                              int k0, int n0, __bf16 (*Bs)[32]) {
  int t = threadIdx.x;
  int n = t >> 2;               // 0..63
  int kc = (t & 3) * 8;
  const float* src = B + (size_t)(n0 + n) * ldb + k0 + kc;
  v4f a = *(const v4f*)src;
  v4f b = *(const v4f*)(src + 4);
  v8bf o;
#pragma unroll
  for (int i = 0; i < 4; ++i) { o[i] = (__bf16)a[i]; o[4 + i] = (__bf16)b[i]; }
  *(v8bf*)&Bs[n][kc] = o;
}

// Wave-level MMA over one 32-deep K tile: batch all fragment loads, then
// issue the 4 WMMAs back-to-back.
__device__ __forceinline__ void wave_mma(const __bf16 (*As)[32],
                                         const __bf16 (*Bs)[32], v8f acc[4]) {
  int lane = threadIdx.x & 31;
  int wave = threadIdx.x >> 5;
  int row = wave * 16 + (lane & 15);
  int kb = (lane >> 4) * 16;      // A: K 0..15 / 16..31 contiguous
  int kh = (lane >> 4) * 16;      // B: K 0..15 / 16..31 contiguous
  int nl = lane & 15;
  v16bf a  = ld_frag(&As[row][kb], &As[row][kb + 8]);
  v16bf b0 = ld_frag(&Bs[nl][kh],      &Bs[nl][kh + 8]);
  v16bf b1 = ld_frag(&Bs[16 + nl][kh], &Bs[16 + nl][kh + 8]);
  v16bf b2 = ld_frag(&Bs[32 + nl][kh], &Bs[32 + nl][kh + 8]);
  v16bf b3 = ld_frag(&Bs[48 + nl][kh], &Bs[48 + nl][kh + 8]);
  acc[0] = wmma_bf16(a, b0, acc[0]);
  acc[1] = wmma_bf16(a, b1, acc[1]);
  acc[2] = wmma_bf16(a, b2, acc[2]);
  acc[3] = wmma_bf16(a, b3, acc[3]);
}

template <bool TRANSB>
__device__ __forceinline__ void gemm_core(const __bf16* __restrict__ A, int lda,
                                          const float* __restrict__ B, int ldb,
                                          int m0, int n0, int K,
                                          __bf16 (*As)[128][32],
                                          __bf16 (*Bs)[64][32],
                                          v8f acc[4]) {
  // prologue: stage tile 0 into buffer 0 (A async, B converting path)
  load_a_tile(A, lda, m0, 0, As[0]);
  if (TRANSB) load_b_tile_t(B, ldb, 0, n0, Bs[0]);
  else        load_b_tile_n(B, ldb, 0, n0, Bs[0]);

  int nsteps = K / 32;
  for (int s = 0; s < nsteps; ++s) {
    int cur = s & 1;
    async_fence();                       // async A-copies for buffer[cur] done
    __syncthreads();                     // buffer[cur] ready for all waves
    if (s + 1 < nsteps) {
      int k1 = (s + 1) * 32;
      if (s + 2 < nsteps) {              // hint tile s+2 toward GL2
        if (TRANSB) __builtin_prefetch(B + (size_t)n0 * ldb + k1 + 32, 0, 1);
        else        __builtin_prefetch(B + (size_t)(k1 + 32) * ldb + n0, 0, 1);
      }
      load_a_tile(A, lda, m0, k1, As[1 - cur]);   // async, flies during MMAs
      if (TRANSB) load_b_tile_t(B, ldb, k1, n0, Bs[1 - cur]);
      else        load_b_tile_n(B, ldb, k1, n0, Bs[1 - cur]);
    }
    wave_mma(As[cur], Bs[cur], acc);
  }
}

// C-fragment element position: row = mbase + r, col = nbase + j*16
__device__ __forceinline__ int c_mbase(int m0) {
  int lane = threadIdx.x & 31, wave = threadIdx.x >> 5;
  return m0 + wave * 16 + ((lane >> 4) ? 8 : 0);
}
__device__ __forceinline__ int c_nbase(int n0) {
  return n0 + (threadIdx.x & 15);
}

// ---------------- block reductions ----------------
__device__ __forceinline__ float block_reduce_sum(float v, float* red) {
  int t = threadIdx.x;
  __syncthreads();
  red[t] = v;
  __syncthreads();
#pragma unroll
  for (int s = 128; s > 0; s >>= 1) {
    if (t < s) red[t] += red[t + s];
    __syncthreads();
  }
  return red[0];
}

// ---------------- kernel 1: x+rotary+pos, LayerNorm -> bf16 ----------------
__global__ __launch_bounds__(256) void ln1_kernel(
    const float* __restrict__ x, const float* __restrict__ pos_emb,
    const float* __restrict__ g, const float* __restrict__ b,
    __bf16* __restrict__ h1) {
  __shared__ float red[256];
  int row = blockIdx.x;           // 0..NTOK-1
  int tt = row % SEQ;
  int t = threadIdx.x;

  float vals[4];
  float s = 0.f;
#pragma unroll
  for (int i = 0; i < 4; ++i) {
    int d = i * 256 + t;
    int j = d >> 1;
    float ang = TWO_PI * (float)tt * ((float)j * (1.0f / 512.0f));
    float rot = (d & 1) ? __cosf(ang) : __sinf(ang);
    float v = x[(size_t)row * DMODEL + d] + rot + pos_emb[(size_t)tt * DMODEL + d];
    vals[i] = v;
    s += v;
  }
  float mu = block_reduce_sum(s, red) * (1.0f / DMODEL);
  float vs = 0.f;
#pragma unroll
  for (int i = 0; i < 4; ++i) {
    float d = vals[i] - mu;
    vs += d * d;
  }
  float var = block_reduce_sum(vs, red) * (1.0f / DMODEL);
  float rstd = rsqrtf(var + EPS);
#pragma unroll
  for (int i = 0; i < 4; ++i) {
    int d = i * 256 + t;
    float o = (vals[i] - mu) * rstd * g[d] + b[d];
    h1[(size_t)row * DMODEL + d] = (__bf16)o;
  }
}

// ---------------- kernel 2: QKV projections ----------------
__global__ __launch_bounds__(256) void gemm_qkv_kernel(
    const __bf16* __restrict__ h1, const float* __restrict__ wq,
    const float* __restrict__ wk, const float* __restrict__ wv,
    __bf16* __restrict__ qb, __bf16* __restrict__ kb, __bf16* __restrict__ vb) {
  __shared__ __bf16 As[2][128][32];
  __shared__ __bf16 Bs[2][64][32];
  int m0 = blockIdx.x * 128;
  int head = blockIdx.y;
  int which = blockIdx.z;
  const float* W = (which == 0 ? wq : which == 1 ? wk : wv) +
                   (size_t)head * DMODEL * HEADD;
  __bf16* Out = (which == 0 ? qb : which == 1 ? kb : vb);

  v8f acc[4] = {};
  gemm_core<false>(h1, DMODEL, W, HEADD, m0, 0, DMODEL, As, Bs, acc);

  int mb = c_mbase(m0);
  int nb = head * HEADD + c_nbase(0);
#pragma unroll
  for (int j = 0; j < 4; ++j)
#pragma unroll
    for (int r = 0; r < 8; ++r)
      Out[(size_t)(mb + r) * DMODEL + nb + j * 16] = (__bf16)acc[j][r];
}

// ---------------- kernel 3: flash attention (causal) ----------------
__global__ __launch_bounds__(256) void attn_kernel(
    const __bf16* __restrict__ Q, const __bf16* __restrict__ K,
    const __bf16* __restrict__ V, __bf16* __restrict__ O) {
  // grid: (SEQ/128, BATCH*NHEAD)
  __shared__ __bf16 Qs[128][64];        // q rows (bf16)
  __shared__ __bf16 Ks[2][32][64];      // K chunk [s][e] (== B^T layout for S)
  __shared__ __bf16 Vt[2][64][32];      // V chunk transposed [e][s]
  __shared__ __bf16 Ps[8][16][32];      // per-wave P staging

  int qbase = blockIdx.x * 128;
  int bh = blockIdx.y;
  int b = bh / NHEAD, h = bh % NHEAD;
  size_t rowB = (size_t)b * SEQ;
  int col0 = h * HEADD;

  int t = threadIdx.x;
  int lane = t & 31, wave = t >> 5;

  { // load Q block: 128x64 bf16, async DMA (4x b128 per thread)
    int r = t >> 1;
    int c = (t & 1) * 32;
    const __bf16* src = Q + (rowB + qbase + r) * DMODEL + col0 + c;
    cp16_g2l(src,      &Qs[r][c]);
    cp16_g2l(src + 8,  &Qs[r][c + 8]);
    cp16_g2l(src + 16, &Qs[r][c + 16]);
    cp16_g2l(src + 24, &Qs[r][c + 24]);
  }

  auto load_kv = [&](int ch, int buf) {
    int s0 = ch * 32;
    { // K chunk 32x64, natural layout: async DMA (1x b128 per thread)
      int r = t >> 3;
      int c = (t & 7) * 8;
      cp16_g2l(K + (rowB + s0 + r) * DMODEL + col0 + c, &Ks[buf][r][c]);
    }
    { // V chunk transposed: pairs along s -> ds_store_b32 (VGPR path)
      int sp = (t & 15) * 2;
      int ec = (t >> 4) * 4;
      const __bf16* src = V + (rowB + s0 + sp) * DMODEL + col0 + ec;
      v4bf a = *(const v4bf*)src;
      v4bf c2 = *(const v4bf*)(src + DMODEL);
#pragma unroll
      for (int i = 0; i < 4; ++i) {
        v2bf p;
        p[0] = a[i];
        p[1] = c2[i];
        *(v2bf*)&Vt[buf][ec + i][sp] = p;
      }
    }
  };

  v8f acc[4] = {};
  float mrow[8], lrow[8];
#pragma unroll
  for (int r = 0; r < 8; ++r) { mrow[r] = -__builtin_inff(); lrow[r] = 0.f; }

  const float scale = 0.03125f;   // 1/sqrt(1024)
  int half8 = (lane >> 4) ? 8 : 0;
  int qrow0 = qbase + wave * 16 + half8;
  int nl = lane & 15;

  int nchunks = (qbase + 128) / 32;
  load_kv(0, 0);
  for (int ch = 0; ch < nchunks; ++ch) {
    int s0 = ch * 32;
    int cur = ch & 1;
    async_fence();                         // Q + K[cur] async copies done
    __syncthreads();                       // KV[cur] ready for all waves
    if (ch + 1 < nchunks) load_kv(ch + 1, 1 - cur);   // prefetch next chunk

    // S = Q K^T : wave's 16 rows x 32 cols, contraction over e (2 k-steps);
    // batch all fragment loads, then 4 WMMAs.
    v8f s[2] = {};
    {
      int arow = wave * 16 + nl;
      int kb = (lane >> 4) * 16;
      int kh = (lane >> 4) * 16;
      v16bf a0  = ld_frag(&Qs[arow][kb],      &Qs[arow][kb + 8]);
      v16bf a1  = ld_frag(&Qs[arow][32 + kb], &Qs[arow][32 + kb + 8]);
      v16bf b00 = ld_frag(&Ks[cur][nl][kh],            &Ks[cur][nl][kh + 8]);
      v16bf b01 = ld_frag(&Ks[cur][16 + nl][kh],       &Ks[cur][16 + nl][kh + 8]);
      v16bf b10 = ld_frag(&Ks[cur][nl][32 + kh],       &Ks[cur][nl][32 + kh + 8]);
      v16bf b11 = ld_frag(&Ks[cur][16 + nl][32 + kh],  &Ks[cur][16 + nl][32 + kh + 8]);
      s[0] = wmma_bf16(a0, b00, s[0]);
      s[1] = wmma_bf16(a0, b01, s[1]);
      s[0] = wmma_bf16(a1, b10, s[0]);
      s[1] = wmma_bf16(a1, b11, s[1]);
    }

    // scale + causal mask + online softmax (row reduce over 16 lanes)
#pragma unroll
    for (int r = 0; r < 8; ++r) {
      int qg = qrow0 + r;
#pragma unroll
      for (int j = 0; j < 2; ++j) {
        int sg = s0 + j * 16 + nl;
        float v = s[j][r] * scale;
        s[j][r] = (sg <= qg) ? v : -__builtin_inff();
      }
      float mx = fmaxf(s[0][r], s[1][r]);
#pragma unroll
      for (int off = 1; off < 16; off <<= 1)
        mx = fmaxf(mx, __shfl_xor(mx, off, 32));
      float mnew = fmaxf(mrow[r], mx);
      float corr = __expf(mrow[r] - mnew);
      float psum = 0.f;
#pragma unroll
      for (int j = 0; j < 2; ++j) {
        float p = __expf(s[j][r] - mnew);
        s[j][r] = p;
        psum += p;
      }
#pragma unroll
      for (int off = 1; off < 16; off <<= 1)
        psum += __shfl_xor(psum, off, 32);
      lrow[r] = lrow[r] * corr + psum;
      mrow[r] = mnew;
#pragma unroll
      for (int j = 0; j < 4; ++j) acc[j][r] *= corr;
#pragma unroll
      for (int j = 0; j < 2; ++j)
        Ps[wave][r + half8][j * 16 + nl] = (__bf16)s[j][r];
    }
    __syncthreads();   // make Ps visible across lanes of the wave

    // O += P V : A = Ps (16x32 over s), B = Vt[e][s]; batched frags.
    {
      int kb2 = (lane >> 4) * 16;
      int kh2 = (lane >> 4) * 16;
      v16bf a  = ld_frag(&Ps[wave][nl][kb2], &Ps[wave][nl][kb2 + 8]);
      v16bf b0 = ld_frag(&Vt[cur][nl][kh2],      &Vt[cur][nl][kh2 + 8]);
      v16bf b1 = ld_frag(&Vt[cur][16 + nl][kh2], &Vt[cur][16 + nl][kh2 + 8]);
      v16bf b2 = ld_frag(&Vt[cur][32 + nl][kh2], &Vt[cur][32 + nl][kh2 + 8]);
      v16bf b3 = ld_frag(&Vt[cur][48 + nl][kh2], &Vt[cur][48 + nl][kh2 + 8]);
      acc[0] = wmma_bf16(a, b0, acc[0]);
      acc[1] = wmma_bf16(a, b1, acc[1]);
      acc[2] = wmma_bf16(a, b2, acc[2]);
      acc[3] = wmma_bf16(a, b3, acc[3]);
    }
  }

  // epilogue: normalize and store [b, t, h, e] (already the concat layout)
#pragma unroll
  for (int r = 0; r < 8; ++r) {
    float inv = 1.0f / lrow[r];
#pragma unroll
    for (int j = 0; j < 4; ++j) {
      int e = j * 16 + nl;
      O[(rowB + qrow0 + r) * DMODEL + col0 + e] = (__bf16)(acc[j][r] * inv);
    }
  }
}

// ---------------- kernel 4: out projection + bias + residual ----------------
__global__ __launch_bounds__(256) void gemm_proj_kernel(
    const __bf16* __restrict__ attn, const float* __restrict__ wproj,
    const float* __restrict__ bproj, const float* __restrict__ x,
    float* __restrict__ x2, float* __restrict__ out) {
  __shared__ __bf16 As[2][128][32];
  __shared__ __bf16 Bs[2][64][32];
  int m0 = blockIdx.x * 128;
  int n0 = blockIdx.y * 64;

  v8f acc[4] = {};
  gemm_core<true>(attn, DMODEL, wproj, DMODEL, m0, n0, DMODEL, As, Bs, acc);

  int mb = c_mbase(m0);
  int nb = c_nbase(n0);
#pragma unroll
  for (int j = 0; j < 4; ++j) {
    int n = nb + j * 16;
    float bias = bproj[n];
#pragma unroll
    for (int r = 0; r < 8; ++r) {
      size_t idx = (size_t)(mb + r) * DMODEL + n;
      float v = x[idx] + acc[j][r] + bias;
      x2[idx] = v;
      out[idx] = v;     // residual seed; experts accumulate on top
    }
  }
}

// ---------------- kernel 5: LN2 + gate logits + top-2 softmax ----------------
__global__ __launch_bounds__(256) void ln2_gate_kernel(
    const float* __restrict__ x2, const float* __restrict__ g,
    const float* __restrict__ b, const float* __restrict__ gate_w,
    __bf16* __restrict__ h2, float* __restrict__ cw) {
  __shared__ float red[256];
  __shared__ float logits[NEXP];
  int row = blockIdx.x;
  int t = threadIdx.x;

  float vals[4];
  float s = 0.f;
#pragma unroll
  for (int i = 0; i < 4; ++i) {
    float v = x2[(size_t)row * DMODEL + i * 256 + t];
    vals[i] = v;
    s += v;
  }
  float mu = block_reduce_sum(s, red) * (1.0f / DMODEL);
  float vs = 0.f;
#pragma unroll
  for (int i = 0; i < 4; ++i) { float d = vals[i] - mu; vs += d * d; }
  float var = block_reduce_sum(vs, red) * (1.0f / DMODEL);
  float rstd = rsqrtf(var + EPS);

  float pe[NEXP] = {};
#pragma unroll
  for (int i = 0; i < 4; ++i) {
    int d = i * 256 + t;
    float o = (vals[i] - mu) * rstd * g[d] + b[d];
    h2[(size_t)row * DMODEL + d] = (__bf16)o;
#pragma unroll
    for (int e = 0; e < NEXP; ++e) pe[e] += o * gate_w[(size_t)e * DMODEL + d];
  }
#pragma unroll
  for (int e = 0; e < NEXP; ++e) {
    float tot = block_reduce_sum(pe[e], red);
    if (t == 0) logits[e] = tot;
  }
  __syncthreads();
  if (t == 0) {
    int e0 = 0;
    for (int e = 1; e < NEXP; ++e) if (logits[e] > logits[e0]) e0 = e;
    int e1 = (e0 == 0) ? 1 : 0;
    for (int e = 0; e < NEXP; ++e)
      if (e != e0 && logits[e] > logits[e1]) e1 = e;
    float ex = __expf(logits[e1] - logits[e0]);
    float w0 = 1.0f / (1.0f + ex);
    float w1 = ex / (1.0f + ex);
    for (int e = 0; e < NEXP; ++e) cw[(size_t)row * NEXP + e] = 0.f;
    cw[(size_t)row * NEXP + e0] = w0;
    cw[(size_t)row * NEXP + e1] = w1;
  }
}

// ---------------- MoE expert GEMMs with tile skipping ----------------
__device__ __forceinline__ bool tile_active(const float* cw, int m0, int expert,
                                            int* s_act) {
  if (threadIdx.x == 0) *s_act = 0;
  __syncthreads();
  if (threadIdx.x < 128 &&
      cw[(size_t)(m0 + threadIdx.x) * NEXP + expert] != 0.0f)
    atomicOr(s_act, 1);
  __syncthreads();
  return *s_act != 0;
}

__global__ __launch_bounds__(256) void gemm_moe1_kernel(
    const __bf16* __restrict__ h2, const float* __restrict__ e_w1,
    const float* __restrict__ e_b1, const float* __restrict__ cw,
    int expert, __bf16* __restrict__ hid) {
  __shared__ __bf16 As[2][128][32];
  __shared__ __bf16 Bs[2][64][32];
  __shared__ int s_act;
  int m0 = blockIdx.x * 128;
  int n0 = blockIdx.y * 64;
  if (!tile_active(cw, m0, expert, &s_act)) return;

  const float* W = e_w1 + (size_t)expert * DMODEL * FDIM;
  const float* bias = e_b1 + (size_t)expert * FDIM;

  v8f acc[4] = {};
  gemm_core<false>(h2, DMODEL, W, FDIM, m0, n0, DMODEL, As, Bs, acc);

  int mb = c_mbase(m0);
  int nb = c_nbase(n0);
#pragma unroll
  for (int j = 0; j < 4; ++j) {
    int n = nb + j * 16;
    float bv = bias[n];
#pragma unroll
    for (int r = 0; r < 8; ++r) {
      float v = fmaxf(acc[j][r] + bv, 0.0f);    // ReLU
      hid[(size_t)(mb + r) * FDIM + n] = (__bf16)v;
    }
  }
}

__global__ __launch_bounds__(256) void gemm_moe2_kernel(
    const __bf16* __restrict__ hid, const float* __restrict__ e_w2,
    const float* __restrict__ e_b2, const float* __restrict__ cw,
    int expert, float* __restrict__ out) {
  __shared__ __bf16 As[2][128][32];
  __shared__ __bf16 Bs[2][64][32];
  __shared__ int s_act;
  int m0 = blockIdx.x * 128;
  int n0 = blockIdx.y * 64;
  if (!tile_active(cw, m0, expert, &s_act)) return;

  const float* W = e_w2 + (size_t)expert * FDIM * DMODEL;
  const float* bias = e_b2 + (size_t)expert * DMODEL;

  v8f acc[4] = {};
  gemm_core<false>(hid, FDIM, W, DMODEL, m0, n0, FDIM, As, Bs, acc);

  int mb = c_mbase(m0);
  int nb = c_nbase(n0);
#pragma unroll
  for (int j = 0; j < 4; ++j) {
    int n = nb + j * 16;
    float bv = bias[n];
#pragma unroll
    for (int r = 0; r < 8; ++r) {
      float sc = cw[(size_t)(mb + r) * NEXP + expert];
      size_t idx = (size_t)(mb + r) * DMODEL + n;
      out[idx] = out[idx] + sc * (acc[j][r] + bv);
    }
  }
}

// ---------------- launcher ----------------
extern "C" void kernel_launch(void* const* d_in, const int* in_sizes, int n_in,
                              void* d_out, int out_size, void* d_ws, size_t ws_size,
                              hipStream_t stream) {
  (void)in_sizes; (void)n_in; (void)out_size; (void)ws_size;
  const float* x       = (const float*)d_in[0];
  const float* pos_emb = (const float*)d_in[1];
  const float* wq      = (const float*)d_in[2];
  const float* wk      = (const float*)d_in[3];
  const float* wv      = (const float*)d_in[4];
  const float* w_proj  = (const float*)d_in[5];
  const float* b_proj  = (const float*)d_in[6];
  const float* ln1_g   = (const float*)d_in[7];
  const float* ln1_b   = (const float*)d_in[8];
  const float* ln2_g   = (const float*)d_in[9];
  const float* ln2_b   = (const float*)d_in[10];
  const float* gate_w  = (const float*)d_in[11];
  const float* e_w1    = (const float*)d_in[12];
  const float* e_b1    = (const float*)d_in[13];
  const float* e_w2    = (const float*)d_in[14];
  const float* e_b2    = (const float*)d_in[15];
  float* out = (float*)d_out;

  char* ws = (char*)d_ws;
  size_t off = 0;
  auto alloc = [&](size_t bytes) -> void* {
    void* p = ws + off;
    off += (bytes + 255) & ~(size_t)255;
    return p;
  };
  __bf16* h1   = (__bf16*)alloc((size_t)NTOK * DMODEL * 2);
  __bf16* qb   = (__bf16*)alloc((size_t)NTOK * DMODEL * 2);
  __bf16* kb   = (__bf16*)alloc((size_t)NTOK * DMODEL * 2);
  __bf16* vb   = (__bf16*)alloc((size_t)NTOK * DMODEL * 2);
  __bf16* attn = (__bf16*)alloc((size_t)NTOK * DMODEL * 2);
  float*  x2   = (float*) alloc((size_t)NTOK * DMODEL * 4);
  __bf16* h2   = (__bf16*)alloc((size_t)NTOK * DMODEL * 2);
  float*  cw   = (float*) alloc((size_t)NTOK * NEXP * 4);
  __bf16* hid  = (__bf16*)alloc((size_t)NTOK * FDIM * 2);

  ln1_kernel<<<NTOK, 256, 0, stream>>>(x, pos_emb, ln1_g, ln1_b, h1);

  gemm_qkv_kernel<<<dim3(NTOK / 128, NHEAD, 3), 256, 0, stream>>>(
      h1, wq, wk, wv, qb, kb, vb);

  attn_kernel<<<dim3(SEQ / 128, BATCH * NHEAD), 256, 0, stream>>>(qb, kb, vb, attn);

  gemm_proj_kernel<<<dim3(NTOK / 128, DMODEL / 64), 256, 0, stream>>>(
      attn, w_proj, b_proj, x, x2, out);

  ln2_gate_kernel<<<NTOK, 256, 0, stream>>>(x2, ln2_g, ln2_b, gate_w, h2, cw);

  for (int e = 0; e < NEXP; ++e) {
    gemm_moe1_kernel<<<dim3(NTOK / 128, FDIM / 64), 256, 0, stream>>>(
        h2, e_w1, e_b1, cw, e, hid);
    gemm_moe2_kernel<<<dim3(NTOK / 128, DMODEL / 64), 256, 0, stream>>>(
        hid, e_w2, e_b2, cw, e, out);
  }
}
